// MultiHeadAttentionLayer_70317204570347
// MI455X (gfx1250) — matrix operand order repaired and checked by reading, not compiled, not run on hardware
//
#include <hip/hip_runtime.h>
#include <hip/hip_bf16.h>

typedef __attribute__((ext_vector_type(16))) _Float16 v16h;
typedef __attribute__((ext_vector_type(8)))  _Float16 h8v;
typedef __attribute__((ext_vector_type(4)))  _Float16 h4v;
typedef __attribute__((ext_vector_type(8)))  float    v8f;

#define NH   16
#define SEQ  2048
#define DMOD 1024
#define NEG_INF (-1.0e30f)

static __device__ __forceinline__ v8f wmma16(v16h a, v16h b, v8f c) {
  return __builtin_amdgcn_wmma_f32_16x16x32_f16(false, a, false, b, (short)0, c,
                                                false, false);
}

// A fragment: 16xK row-major source. Per-lane data = two contiguous 16B runs:
// a[0..7] = row[0..7], a[8..15] = row[16..23]  (row = base + (L&15)*stride +
// koff + (L>>4)*8). Emits 2x ds_load_b128 when 16B aligned.
static __device__ __forceinline__ v16h frag_a(const _Float16* p, int lane,
                                              int stride, int koff) {
  const _Float16* row = p + (lane & 15) * stride + koff + ((lane >> 4) & 1) * 8;
  h8v lo = *(const h8v*)(row);
  h8v hi = *(const h8v*)(row + 16);
  v16h a;
#pragma unroll
  for (int j = 0; j < 8; ++j) { a[j] = lo[j]; a[8 + j] = hi[j]; }
  return a;
}

// B fragment where the SOURCE holds B transposed, i.e. src[n][k] row-major.
// Per-lane data = two contiguous 16B runs: b[0..7] = row[0..7], b[8..15] =
// row[16..23], row = base + (noff + (L&15))*stride + koff + (L>>4)*8. Emits 2x ds_load_b128.
static __device__ __forceinline__ v16h frag_bT(const _Float16* p, int lane,
                                               int stride, int koff, int noff) {
  const _Float16* row =
      p + (size_t)(noff + (lane & 15)) * stride + koff + ((lane >> 4) & 1) * 8;
  h8v lo = *(const h8v*)(row);
  h8v hi = *(const h8v*)(row + 16);
  v16h b;
#pragma unroll
  for (int j = 0; j < 8; ++j) { b[j] = lo[j]; b[8 + j] = hi[j]; }
  return b;
}

// ---------------------------------------------------------------------------
// GEMM 1: C = X(4096x1024,f32) * W(1024x1024,f32) + bias -> f16 head-major
// dst[((b*16+head)*2048 + s)*64 + dk]. 256 thr / 8 waves, tile 128x128, BK=32.
// W tile stored TRANSPOSED in LDS so B fragments are contiguous b128 loads.
// ---------------------------------------------------------------------------
__global__ __launch_bounds__(256) void gemm_proj_kernel(
    const float* __restrict__ X, const float* __restrict__ W,
    const float* __restrict__ bias, _Float16* __restrict__ dst) {
  __shared__ __align__(16) _Float16 As[128 * 40];   // [m][k] 128x32 (+pad)
  __shared__ __align__(16) _Float16 BsT[128 * 40];  // [n][k] 128x32 (+pad)
  const int t = threadIdx.x;
  const int lane = t & 31, w = t >> 5;
  const int wr = w >> 1, wc = w & 1;
  const int m0 = blockIdx.y * 128;
  const int n0 = blockIdx.x * 128;

  v8f acc[2][4];
#pragma unroll
  for (int mt = 0; mt < 2; ++mt)
#pragma unroll
    for (int nt = 0; nt < 4; ++nt) acc[mt][nt] = v8f{0, 0, 0, 0, 0, 0, 0, 0};

  for (int k0 = 0; k0 < DMOD; k0 += 32) {
    __syncthreads();
#pragma unroll
    for (int i = 0; i < 4; ++i) {  // A tile 128x32 f32, float4 per thread
      int e = t + i * 256;
      int r = e >> 3, c = (e & 7) * 4;
      const float4 xv = *(const float4*)&X[(size_t)(m0 + r) * DMOD + k0 + c];
      h4v hv = {(_Float16)xv.x, (_Float16)xv.y, (_Float16)xv.z, (_Float16)xv.w};
      *(h4v*)&As[r * 40 + c] = hv;  // 8B aligned store
    }
#pragma unroll
    for (int i = 0; i < 4; ++i) {  // B tile 32x128 f32 -> transposed LDS
      int e = t + i * 256;
      int r = e >> 5, c4 = (e & 31) * 4;
      const float4 wv = *(const float4*)&W[(size_t)(k0 + r) * DMOD + n0 + c4];
      BsT[(c4 + 0) * 40 + r] = (_Float16)wv.x;
      BsT[(c4 + 1) * 40 + r] = (_Float16)wv.y;
      BsT[(c4 + 2) * 40 + r] = (_Float16)wv.z;
      BsT[(c4 + 3) * 40 + r] = (_Float16)wv.w;
    }
    __syncthreads();

    v16h af[2], bf[4];
#pragma unroll
    for (int mt = 0; mt < 2; ++mt)
      af[mt] = frag_a(As + (wr * 32 + mt * 16) * 40, lane, 40, 0);
#pragma unroll
    for (int nt = 0; nt < 4; ++nt)
      bf[nt] = frag_bT(BsT, lane, 40, 0, wc * 64 + nt * 16);
#pragma unroll
    for (int mt = 0; mt < 2; ++mt)
#pragma unroll
      for (int nt = 0; nt < 4; ++nt)
        acc[mt][nt] = wmma16(af[mt], bf[nt], acc[mt][nt]);
  }

  const int n = lane & 15;
  const int h8 = (lane >> 4) * 8;
#pragma unroll
  for (int mt = 0; mt < 2; ++mt)
#pragma unroll
    for (int nt = 0; nt < 4; ++nt) {
      int col = n0 + wc * 64 + nt * 16 + n;
      float bcol = bias[col];
      int head = col >> 6, dk = col & 63;
#pragma unroll
      for (int r = 0; r < 8; ++r) {
        int row = m0 + wr * 32 + mt * 16 + h8 + r;
        int bb = row >> 11, s = row & 2047;
        float v = acc[mt][nt][r] + bcol;
        dst[(((size_t)bb * NH + head) * SEQ + s) * 64 + dk] = (_Float16)v;
      }
    }
}

// ---------------------------------------------------------------------------
// Flash attention: grid.x = query block (64 q), grid.y = b*H + head.
// 4 waves x 16 queries; shared 64-key K/V LDS tiles. K kept row-major
// (QK^T B-fragments are already contiguous); V stored TRANSPOSED [dv][key].
// ---------------------------------------------------------------------------
__global__ __launch_bounds__(128) void attn_kernel(
    const _Float16* __restrict__ Q, const _Float16* __restrict__ K,
    const _Float16* __restrict__ V, _Float16* __restrict__ OW) {
  __shared__ __align__(16) _Float16 Kt[64 * 72];   // [key][dk]
  __shared__ __align__(16) _Float16 VtT[64 * 72];  // [dv][key]
  __shared__ __align__(16) _Float16 Ps[4][16 * 72];

  const int bh = blockIdx.y;
  const int b = bh >> 4, head = bh & 15;
  const _Float16* Qh = Q + (size_t)bh * SEQ * 64;
  const _Float16* Kh = K + (size_t)bh * SEQ * 64;
  const _Float16* Vh = V + (size_t)bh * SEQ * 64;

  const int w = threadIdx.x >> 5;
  const int lane = threadIdx.x & 31;
  const int n = lane & 15;
  const int h8 = (lane >> 4) * 8;
  const int q0 = blockIdx.x * 64 + w * 16;

  // Q fragments (dk 0..31, 32..63), pre-scaled by 1/sqrt(64) = 0.125 (exact).
  v16h qa[2];
  {
    const _Float16 s8 = (_Float16)0.125f;
    const _Float16* qrow = Qh + (size_t)(q0 + n) * 64 + ((lane >> 4) & 1) * 8;
#pragma unroll
    for (int kt = 0; kt < 2; ++kt) {
      h8v lo = *(const h8v*)(qrow + kt * 32);
      h8v hi = *(const h8v*)(qrow + kt * 32 + 16);
#pragma unroll
      for (int j = 0; j < 8; ++j) {
        qa[kt][j] = lo[j] * s8;
        qa[kt][8 + j] = hi[j] * s8;
      }
    }
  }

  float m_s[8], l_s[8];
  v8f oacc[4];
#pragma unroll
  for (int r = 0; r < 8; ++r) { m_s[r] = NEG_INF; l_s[r] = 0.f; }
#pragma unroll
  for (int nt = 0; nt < 4; ++nt) oacc[nt] = v8f{0, 0, 0, 0, 0, 0, 0, 0};

  const int nkv = blockIdx.x + 1;  // uniform trip count across the 4 waves
  for (int kv = 0; kv < nkv; ++kv) {
    const int kv0 = kv * 64;
    __syncthreads();
#pragma unroll
    for (int i = 0; i < 4; ++i) {  // 128x32 halfs, 16B chunks, 128 threads
      int e = threadIdx.x + i * 128;
      int row = e >> 2;
      int c8 = (e & 3) * 8;
      // K: straight copy (row-major)
      *(h8v*)&Kt[row * 72 + c8] =
          *(const h8v*)&Kh[(size_t)(kv0 + row) * 64 + c8];
      // V: transpose into [dv][key]
      h8v vv = *(const h8v*)&Vh[(size_t)(kv0 + row) * 64 + c8];
#pragma unroll
      for (int ii = 0; ii < 8; ++ii) VtT[(c8 + ii) * 72 + row] = vv[ii];
    }
    if (kv + 1 < nkv) {  // prefetch next K/V tile (global_prefetch_b8)
      const size_t nrow = (size_t)(kv0 + 64 + (threadIdx.x >> 1)) * 64;
      __builtin_prefetch(&Kh[nrow], 0, 0);
      __builtin_prefetch(&Vh[nrow], 0, 0);
    }
    __syncthreads();

    // S = (Q/8) K^T  -> 16x64 per wave
    v8f sc[4];
#pragma unroll
    for (int nt = 0; nt < 4; ++nt) {
      v8f s = v8f{0, 0, 0, 0, 0, 0, 0, 0};
      s = wmma16(qa[0], frag_bT(Kt, lane, 72, 0, nt * 16), s);
      s = wmma16(qa[1], frag_bT(Kt, lane, 72, 32, nt * 16), s);
      sc[nt] = s;
    }

    // |.| + causal mask
#pragma unroll
    for (int nt = 0; nt < 4; ++nt)
#pragma unroll
      for (int r = 0; r < 8; ++r) {
        float vv = fabsf(sc[nt][r]);
        int sk = kv0 + nt * 16 + n;
        int sq = q0 + h8 + r;
        sc[nt][r] = (sk > sq) ? NEG_INF : vv;
      }

    // online softmax (per row; cross-lane reduce within each 16-lane half)
    _Float16* pw = &Ps[w][0];
#pragma unroll
    for (int r = 0; r < 8; ++r) {
      float rm = fmaxf(fmaxf(sc[0][r], sc[1][r]), fmaxf(sc[2][r], sc[3][r]));
      rm = fmaxf(rm, __shfl_xor(rm, 1, 16));
      rm = fmaxf(rm, __shfl_xor(rm, 2, 16));
      rm = fmaxf(rm, __shfl_xor(rm, 4, 16));
      rm = fmaxf(rm, __shfl_xor(rm, 8, 16));
      float mnew = fmaxf(m_s[r], rm);
      float alpha = __expf(m_s[r] - mnew);
      m_s[r] = mnew;
      float rs = 0.f;
#pragma unroll
      for (int nt = 0; nt < 4; ++nt) {
        float pv = __expf(sc[nt][r] - mnew);
        sc[nt][r] = pv;
        rs += pv;
        pw[(h8 + r) * 72 + nt * 16 + n] = (_Float16)pv;
      }
      rs += __shfl_xor(rs, 1, 16);
      rs += __shfl_xor(rs, 2, 16);
      rs += __shfl_xor(rs, 4, 16);
      rs += __shfl_xor(rs, 8, 16);
      l_s[r] = l_s[r] * alpha + rs;
#pragma unroll
      for (int nt = 0; nt < 4; ++nt) oacc[nt][r] *= alpha;
    }

    // O += P V   (P reshaped C->A via per-wave LDS; same-wave DS in-order)
    v16h pa0 = frag_a(pw, lane, 72, 0);
    v16h pa1 = frag_a(pw, lane, 72, 32);
#pragma unroll
    for (int nt = 0; nt < 4; ++nt) {
      oacc[nt] = wmma16(pa0, frag_bT(VtT, lane, 72, 0, nt * 16), oacc[nt]);
      oacc[nt] = wmma16(pa1, frag_bT(VtT, lane, 72, 32, nt * 16), oacc[nt]);
    }
  }

  // normalize, write token-major f16 (row = b*S + sq, col = head*64 + dv)
#pragma unroll
  for (int nt = 0; nt < 4; ++nt) {
    float inv;
#pragma unroll
    for (int r = 0; r < 8; ++r) {
      inv = 1.0f / l_s[r];
      float ov = oacc[nt][r] * inv;
      int sq = q0 + h8 + r;
      int col = head * 64 + nt * 16 + n;
      OW[(size_t)(b * SEQ + sq) * DMOD + col] = (_Float16)ov;
    }
  }
}

// ---------------------------------------------------------------------------
// GEMM 2: out = O(4096x1024,f16) * Wo(1024x1024,f32->f16) + bo, fp32 output.
// ---------------------------------------------------------------------------
__global__ __launch_bounds__(256) void gemm_out_kernel(
    const _Float16* __restrict__ A, const float* __restrict__ W,
    const float* __restrict__ bias, float* __restrict__ out) {
  __shared__ __align__(16) _Float16 As[128 * 40];
  __shared__ __align__(16) _Float16 BsT[128 * 40];
  const int t = threadIdx.x;
  const int lane = t & 31, w = t >> 5;
  const int wr = w >> 1, wc = w & 1;
  const int m0 = blockIdx.y * 128;
  const int n0 = blockIdx.x * 128;

  v8f acc[2][4];
#pragma unroll
  for (int mt = 0; mt < 2; ++mt)
#pragma unroll
    for (int nt = 0; nt < 4; ++nt) acc[mt][nt] = v8f{0, 0, 0, 0, 0, 0, 0, 0};

  for (int k0 = 0; k0 < DMOD; k0 += 32) {
    __syncthreads();
#pragma unroll
    for (int i = 0; i < 2; ++i) {  // A tile 128x32 f16, 16B chunks
      int e = t + i * 256;
      int r = e >> 2, c8 = (e & 3) * 8;
      *(h8v*)&As[r * 40 + c8] =
          *(const h8v*)&A[(size_t)(m0 + r) * DMOD + k0 + c8];
    }
#pragma unroll
    for (int i = 0; i < 4; ++i) {  // B tile 32x128 f32 -> transposed LDS
      int e = t + i * 256;
      int r = e >> 5, c4 = (e & 31) * 4;
      const float4 wv = *(const float4*)&W[(size_t)(k0 + r) * DMOD + n0 + c4];
      BsT[(c4 + 0) * 40 + r] = (_Float16)wv.x;
      BsT[(c4 + 1) * 40 + r] = (_Float16)wv.y;
      BsT[(c4 + 2) * 40 + r] = (_Float16)wv.z;
      BsT[(c4 + 3) * 40 + r] = (_Float16)wv.w;
    }
    __syncthreads();

    v16h af[2], bf[4];
#pragma unroll
    for (int mt = 0; mt < 2; ++mt)
      af[mt] = frag_a(As + (wr * 32 + mt * 16) * 40, lane, 40, 0);
#pragma unroll
    for (int nt = 0; nt < 4; ++nt)
      bf[nt] = frag_bT(BsT, lane, 40, 0, wc * 64 + nt * 16);
#pragma unroll
    for (int mt = 0; mt < 2; ++mt)
#pragma unroll
      for (int nt = 0; nt < 4; ++nt)
        acc[mt][nt] = wmma16(af[mt], bf[nt], acc[mt][nt]);
  }

  const int n = lane & 15;
  const int h8 = (lane >> 4) * 8;
#pragma unroll
  for (int mt = 0; mt < 2; ++mt)
#pragma unroll
    for (int nt = 0; nt < 4; ++nt) {
      int col = n0 + wc * 64 + nt * 16 + n;
      float bcol = bias[col];
#pragma unroll
      for (int r = 0; r < 8; ++r) {
        int row = m0 + wr * 32 + mt * 16 + h8 + r;
        out[(size_t)row * DMOD + col] = acc[mt][nt][r] + bcol;
      }
    }
}

extern "C" void kernel_launch(void* const* d_in, const int* in_sizes, int n_in,
                              void* d_out, int out_size, void* d_ws,
                              size_t ws_size, hipStream_t stream) {
  (void)in_sizes; (void)n_in; (void)out_size; (void)ws_size;
  const float* x  = (const float*)d_in[0];
  const float* Wq = (const float*)d_in[1];
  const float* bq = (const float*)d_in[2];
  const float* Wk = (const float*)d_in[3];
  const float* bk = (const float*)d_in[4];
  const float* Wv = (const float*)d_in[5];
  const float* bv = (const float*)d_in[6];
  const float* Wo = (const float*)d_in[7];
  const float* bo = (const float*)d_in[8];
  float* out = (float*)d_out;

  const size_t QKV = (size_t)2 * NH * SEQ * 64;  // halfs per tensor
  _Float16* q  = (_Float16*)d_ws;
  _Float16* k  = q + QKV;
  _Float16* v  = k + QKV;
  _Float16* ow = v + QKV;

  dim3 gemm_grid(DMOD / 128, (2 * SEQ) / 128, 1);  // (8, 32)
  gemm_proj_kernel<<<gemm_grid, 256, 0, stream>>>(x, Wq, bq, q);
  gemm_proj_kernel<<<gemm_grid, 256, 0, stream>>>(x, Wk, bk, k);
  gemm_proj_kernel<<<gemm_grid, 256, 0, stream>>>(x, Wv, bv, v);

  dim3 attn_grid(SEQ / 64, 2 * NH, 1);  // (32, 32)
  attn_kernel<<<attn_grid, 128, 0, stream>>>(q, k, v, ow);

  gemm_out_kernel<<<gemm_grid, 256, 0, stream>>>(ow, Wo, bo, out);
}
